// PairwiseLoss_2035814498850
// MI455X (gfx1250) — matrix operand order repaired and checked by reading, not compiled, not run on hardware
//
#include <hip/hip_runtime.h>

// ---------------------------------------------------------------------------
// PairwiseLoss for MI455X (gfx1250, wave32).
// S = F @ Pt  via split-precision bf16 WMMA (hi/lo decomposition, 3 products),
// then d2 = max(a2 + b2 - 2S, 0); z = d2 + (B - TAO); clamped softplus; mean.
// Compute-bound: ~8.4 GFLOP GEMM over ~17 MB of input (AI ~485 FLOP/B).
// ---------------------------------------------------------------------------

typedef __attribute__((ext_vector_type(16))) __bf16 v16bf;
typedef __attribute__((ext_vector_type(8)))  __bf16 v8bf;
typedef __attribute__((ext_vector_type(8)))  float  v8f;

#define QROWS 32768
#define CDIM  1000
#define CPAD  1024
#define DDIM  128
#define MT    2      // M-tiles (16 rows) per wave
#define WAVES 4      // waves per block

__device__ __forceinline__ __bf16 bf16_rne(float f) {
    unsigned u = __builtin_bit_cast(unsigned, f);
    unsigned r = u + 0x7FFFu + ((u >> 16) & 1u);   // round-to-nearest-even
    unsigned short h = (unsigned short)(r >> 16);
    return __builtin_bit_cast(__bf16, h);
}

__device__ __forceinline__ float bf16_to_f32(__bf16 h) {
    unsigned short s = __builtin_bit_cast(unsigned short, h);
    unsigned u = ((unsigned)s) << 16;
    return __builtin_bit_cast(float, u);
}

// ---------------------------------------------------------------------------
// Prep: prototypes f32[1000,128] -> padded bf16 hi/lo [1024,128] + b2 norms.
// Also zeroes the global loss accumulator.
// ---------------------------------------------------------------------------
__global__ void prep_kernel(const float* __restrict__ proto,
                            __bf16* __restrict__ pHi,
                            __bf16* __restrict__ pLo,
                            float*  __restrict__ b2,
                            float*  __restrict__ acc) {
    const int j = blockIdx.x;    // 0..1023 (padded prototype row)
    const int t = threadIdx.x;   // 0..127  (feature dim)
    float f = (j < CDIM) ? proto[j * DDIM + t] : 0.0f;
    __bf16 hi = bf16_rne(f);
    __bf16 lo = bf16_rne(f - bf16_to_f32(hi));
    pHi[j * DDIM + t] = hi;
    pLo[j * DDIM + t] = lo;

    float sq = f * f;
    #pragma unroll
    for (int o = 16; o > 0; o >>= 1) sq += __shfl_xor(sq, o, 32);
    __shared__ float red[WAVES];
    if ((t & 31) == 0) red[t >> 5] = sq;
    __syncthreads();
    if (t == 0) {
        b2[j] = red[0] + red[1] + red[2] + red[3];
        if (j == 0) *acc = 0.0f;
    }
}

// ---------------------------------------------------------------------------
// Main: each wave holds 2 M-tiles of A (hi/lo bf16, full K=128) in registers
// and sweeps all 64 N-tiles. 3 WMMAs per K=32 chunk (hi*hi + hi*lo + lo*hi).
// ---------------------------------------------------------------------------
__global__ void __launch_bounds__(WAVES * 32, 1)
gemm_loss_kernel(const float* __restrict__ feat,
                 const __bf16* __restrict__ pHi,
                 const __bf16* __restrict__ pLo,
                 const float* __restrict__ b2,
                 float* __restrict__ acc) {
    const int lane  = threadIdx.x & 31;
    const int wave  = threadIdx.x >> 5;
    const int wtile = blockIdx.x * WAVES + wave;   // 0..1023
    const int half  = lane >> 4;                   // 0 | 1
    const int l15   = lane & 15;
    const int aoff  = half ? 8 : 0;                // A-matrix lane K sub-offset

    // -------- Load + split-convert A tiles (kept in registers) --------
    v16bf aHi[MT][4], aLo[MT][4];
    float a2r[MT][8];

    #pragma unroll
    for (int mt = 0; mt < MT; ++mt) {
        const size_t row = (size_t)((wtile * MT + mt) * 16 + l15);
        const float* fr = feat + row * DDIM;
        float a2lane = 0.0f;
        #pragma unroll
        for (int kc = 0; kc < 4; ++kc) {
            const int kb = kc * 32;
            float v[16];
            const float4 x0 = *(const float4*)(fr + kb + aoff);
            const float4 x1 = *(const float4*)(fr + kb + aoff + 4);
            const float4 x2 = *(const float4*)(fr + kb + 16 + aoff);
            const float4 x3 = *(const float4*)(fr + kb + 16 + aoff + 4);
            v[0]=x0.x;  v[1]=x0.y;  v[2]=x0.z;  v[3]=x0.w;
            v[4]=x1.x;  v[5]=x1.y;  v[6]=x1.z;  v[7]=x1.w;
            v[8]=x2.x;  v[9]=x2.y;  v[10]=x2.z; v[11]=x2.w;
            v[12]=x3.x; v[13]=x3.y; v[14]=x3.z; v[15]=x3.w;
            #pragma unroll
            for (int j = 0; j < 16; ++j) {
                a2lane += v[j] * v[j];
                __bf16 hi = bf16_rne(v[j]);
                aHi[mt][kc][j] = hi;
                aLo[mt][kc][j] = bf16_rne(v[j] - bf16_to_f32(hi));
            }
        }
        // row norm: lanes m and m+16 jointly hold row m
        const float s2 = a2lane + __shfl_xor(a2lane, 16, 32);
        // accumulator VGPR r at this lane maps to row M = r + 8*half
        #pragma unroll
        for (int r = 0; r < 8; ++r)
            a2r[mt][r] = __shfl(s2, r + 8 * half, 32);
    }

    // -------- Sweep N-tiles --------
    const int boff = half ? 8 : 0;    // B-matrix lane K sub-offset
    float gsum = 0.0f;

    for (int nt = 0; nt < CPAD / 16; ++nt) {
        const int n = nt * 16 + l15;                 // this lane's column
        const __bf16* bh = pHi + (size_t)n * DDIM + boff;
        const __bf16* bl = pLo + (size_t)n * DDIM + boff;

        v8f accv[MT];
        #pragma unroll
        for (int mt = 0; mt < MT; ++mt)
            accv[mt] = (v8f){0.f, 0.f, 0.f, 0.f, 0.f, 0.f, 0.f, 0.f};

        #pragma unroll
        for (int kc = 0; kc < 4; ++kc) {
            const int kb = kc * 32;
            const v8bf h0 = *(const v8bf*)(bh + kb);
            const v8bf h1 = *(const v8bf*)(bh + kb + 16);
            const v8bf l0 = *(const v8bf*)(bl + kb);
            const v8bf l1 = *(const v8bf*)(bl + kb + 16);
            v16bf bHi, bLo;
            #pragma unroll
            for (int j = 0; j < 8; ++j) {
                bHi[j] = h0[j]; bHi[j + 8] = h1[j];
                bLo[j] = l0[j]; bLo[j + 8] = l1[j];
            }
            #pragma unroll
            for (int mt = 0; mt < MT; ++mt) {
                accv[mt] = __builtin_amdgcn_wmma_f32_16x16x32_bf16(
                    false, aHi[mt][kc], false, bHi, (short)0, accv[mt], false, false);
                accv[mt] = __builtin_amdgcn_wmma_f32_16x16x32_bf16(
                    false, aHi[mt][kc], false, bLo, (short)0, accv[mt], false, false);
                accv[mt] = __builtin_amdgcn_wmma_f32_16x16x32_bf16(
                    false, aLo[mt][kc], false, bHi, (short)0, accv[mt], false, false);
            }
        }

        // -------- epilogue: d2 -> clamped softplus, masked to n < 1000 --------
        const float b2n = b2[n];
        if (n < CDIM) {
            #pragma unroll
            for (int mt = 0; mt < MT; ++mt) {
                #pragma unroll
                for (int r = 0; r < 8; ++r) {
                    const float s  = accv[mt][r];
                    const float d2 = fmaxf(a2r[mt][r] + b2n - 2.0f * s, 0.0f);
                    const float z  = d2 + (1.0f - 1.0f);   // B - TAO == 0
                    const float soft = __logf(1.0f + __expf(0.1f * fminf(z, 10.0f))) * 10.0f;
                    gsum += (z < 10.0f) ? soft : z;
                }
            }
        }
    }

    // -------- wave reduce + global accumulate --------
    #pragma unroll
    for (int o = 16; o > 0; o >>= 1) gsum += __shfl_xor(gsum, o, 32);
    if (lane == 0) atomicAdd(acc, gsum);
}

__global__ void finalize_kernel(const float* __restrict__ acc,
                                float* __restrict__ out) {
    if (threadIdx.x == 0)
        out[0] = acc[0] * (1.0f / ((float)QROWS * (float)CDIM));
}

// ---------------------------------------------------------------------------
extern "C" void kernel_launch(void* const* d_in, const int* in_sizes, int n_in,
                              void* d_out, int out_size, void* d_ws, size_t ws_size,
                              hipStream_t stream) {
    const float* feat  = (const float*)d_in[0];   // [32768,128] f32
    // d_in[1] = labels (int64) — mathematically dead in the reference
    const float* proto = (const float*)d_in[2];   // [1000,128] f32
    float* out = (float*)d_out;

    char* ws = (char*)d_ws;
    __bf16* pHi = (__bf16*)(ws);
    __bf16* pLo = (__bf16*)(ws + (size_t)CPAD * DDIM * 2);
    float*  b2  = (float*) (ws + (size_t)CPAD * DDIM * 4);
    float*  acc = (float*) (ws + (size_t)CPAD * DDIM * 4 + (size_t)CPAD * 4);

    prep_kernel<<<CPAD, DDIM, 0, stream>>>(proto, pHi, pLo, b2, acc);

    // 2048 M-tiles total: 256 blocks * 4 waves * 2 M-tiles
    gemm_loss_kernel<<<QROWS / (16 * MT * WAVES), WAVES * 32, 0, stream>>>(
        feat, pHi, pLo, b2, acc);

    finalize_kernel<<<1, 32, 0, stream>>>(acc, out);
}
